// PointNetTunable_20117626814615
// MI455X (gfx1250) — matrix operand order repaired and checked by reading, not compiled, not run on hardware
//
#include <hip/hip_runtime.h>

typedef __attribute__((ext_vector_type(16))) _Float16 v16h;
typedef __attribute__((ext_vector_type(8)))  _Float16 v8h;
typedef __attribute__((ext_vector_type(8)))  float    v8f;

#define BATCH 16
#define K_NB  32

// ---------------------------------------------------------------------------
// Farthest point sampling: one block per cloud. Points + running min-dist in
// LDS (dynamic). Matches reference: sampled idx sequence = [0, argmax, ...].
// ---------------------------------------------------------------------------
template<int N, int S, int TPB>
__global__ void __launch_bounds__(TPB)
fps_kernel(const float* __restrict__ xyz, float* __restrict__ out_xyz) {
  extern __shared__ float smem[];
  float* sx = smem;
  float* sy = smem + N;
  float* sz = smem + 2 * N;
  float* dist = smem + 3 * N;
  __shared__ float rv[TPB / 32];
  __shared__ int   ri[TPB / 32];
  __shared__ int   s_last;

  const int b = blockIdx.x;
  const int tid = threadIdx.x;
  const float* p = xyz + (size_t)b * N * 3;

  for (int i = tid; i < N; i += TPB) {
    sx[i] = p[i * 3 + 0];
    sy[i] = p[i * 3 + 1];
    sz[i] = p[i * 3 + 2];
    dist[i] = 1e10f;
  }
  __syncthreads();

  if (tid == 0) {
    out_xyz[((size_t)b * S + 0) * 3 + 0] = sx[0];
    out_xyz[((size_t)b * S + 0) * 3 + 1] = sy[0];
    out_xyz[((size_t)b * S + 0) * 3 + 2] = sz[0];
  }

  int last = 0;
  for (int s = 1; s < S; ++s) {
    const float lx = sx[last], ly = sy[last], lz = sz[last];
    float bv = -1.0f;
    int   bi = 0;
    for (int i = tid; i < N; i += TPB) {
      const float dx = sx[i] - lx, dy = sy[i] - ly, dz = sz[i] - lz;
      const float d = dx * dx + dy * dy + dz * dz;
      const float nd = fminf(dist[i], d);
      dist[i] = nd;
      if (nd > bv) { bv = nd; bi = i; }
    }
    // wave32 argmax reduce, prefer lower index on ties
    #pragma unroll
    for (int off = 16; off >= 1; off >>= 1) {
      const float ov = __shfl_xor(bv, off);
      const int   oi = __shfl_xor(bi, off);
      if (ov > bv || (ov == bv && oi < bi)) { bv = ov; bi = oi; }
    }
    const int lane = tid & 31, wv = tid >> 5;
    if (lane == 0) { rv[wv] = bv; ri[wv] = bi; }
    __syncthreads();
    if (tid == 0) {
      float bbv = rv[0];
      int   bbi = ri[0];
      for (int w = 1; w < TPB / 32; ++w)
        if (rv[w] > bbv) { bbv = rv[w]; bbi = ri[w]; }
      s_last = bbi;
      out_xyz[((size_t)b * S + s) * 3 + 0] = sx[bbi];
      out_xyz[((size_t)b * S + s) * 3 + 1] = sy[bbi];
      out_xyz[((size_t)b * S + s) * 3 + 2] = sz[bbi];
    }
    __syncthreads();
    last = s_last;
  }
}

// ---------------------------------------------------------------------------
// Weight prep: transpose + convert W (CK x COUT, f32) into Wt (COUT x KPAD,
// f16, zero-padded). Makes group_mlp B-fragments contiguous 32-byte loads.
// ---------------------------------------------------------------------------
template<int CK, int KPAD, int COUT>
__global__ void prep_weights_kernel(const float* __restrict__ W,
                                    _Float16* __restrict__ Wt) {
  const int idx = blockIdx.x * blockDim.x + threadIdx.x;
  if (idx < COUT * KPAD) {
    const int col = idx / KPAD, k = idx % KPAD;
    Wt[idx] = (k < CK) ? (_Float16)W[k * COUT + col] : (_Float16)0.0f;
  }
}

// ---------------------------------------------------------------------------
// Fused ball-query + group + MLP (WMMA f16->f32) + k-max. One block per
// (batch, center). COUT/16 waves; wave wt computes output channels
// [wt*16, wt*16+16). k=32 neighbors = two M-tiles (two WMMAs per k-step).
// ---------------------------------------------------------------------------
template<int N, int S, int CIN, int CK, int KPAD, int COUT>
__global__ void __launch_bounds__(COUT * 2)
group_mlp_kernel(const float* __restrict__ xyz, const float* __restrict__ feat,
                 const float* __restrict__ centers,
                 const _Float16* __restrict__ Wt,
                 const float* __restrict__ bias, float* __restrict__ out_feat,
                 float r2) {
  __shared__ int      s_idx[K_NB];
  __shared__ unsigned s_mask[N / 32];
  __shared__ _Float16 s_h[K_NB][KPAD];

  const int bs = blockIdx.x;
  const int b = bs / S, s = bs % S;
  const int tid = threadIdx.x, lane = tid & 31, wv = tid >> 5;
  constexpr int TPB = COUT * 2;

  const float cx = centers[((size_t)(b * S + s)) * 3 + 0];
  const float cy = centers[((size_t)(b * S + s)) * 3 + 1];
  const float cz = centers[((size_t)(b * S + s)) * 3 + 2];
  const float* px = xyz + (size_t)b * N * 3;

  // ---- Phase A1: parallel validity ballots. Wave wv covers the contiguous
  // 32-point chunk [it*TPB + wv*32, +32) each iteration; lane0 stores mask.
  for (int i = tid; i < N; i += TPB) {
    const float dx = px[i * 3 + 0] - cx;
    const float dy = px[i * 3 + 1] - cy;
    const float dz = px[i * 3 + 2] - cz;
    const bool valid = (dx * dx + dy * dy + dz * dz) <= r2;
    const unsigned mm = (unsigned)__ballot(valid);
    if (lane == 0) s_mask[i >> 5] = mm;
  }
  __syncthreads();

  // ---- Phase A2: wave 0 compacts first K_NB valid indices (index order) ----
  if (wv == 0) {
    int cnt = 0;
    for (int c = 0; c < N / 32 && cnt < K_NB; ++c) {
      const unsigned mm = s_mask[c];
      const int pos = cnt + __popc(mm & ((1u << lane) - 1u));
      if (((mm >> lane) & 1u) && pos < K_NB) s_idx[pos] = c * 32 + lane;
      cnt += __popc(mm);
    }
    const int total = cnt < K_NB ? cnt : K_NB; // >= 1 (center itself)
    const int i0 = s_idx[0];
    for (int j = lane; j < K_NB; j += 32)
      if (j >= total) s_idx[j] = i0;
  }
  __syncthreads();

  // ---- Phase B: gather h = [rel_xyz | feat] into LDS as f16 (zero pad) ----
  for (int e = tid; e < K_NB * KPAD; e += TPB) {
    const int m = e / KPAD, k = e % KPAD;
    const int gi = s_idx[m];
    float v = 0.0f;
    if (k < 3) {
      v = px[gi * 3 + k] - ((k == 0) ? cx : (k == 1) ? cy : cz);
    } else if (k < CK) {
      v = feat[((size_t)b * N + gi) * CIN + (k - 3)];
    }
    s_h[m][k] = (_Float16)v;
  }
  __syncthreads();

  // ---- Phase C: WMMA 16x16x32 f16 accumulation over KPAD ----
  v8f acc0 = {};
  v8f acc1 = {};
  const int m0  = lane & 15;
  const int col = (wv << 4) | (lane & 15);
  const _Float16* wrow = Wt + (size_t)col * KPAD;
  for (int kt = 0; kt < KPAD / 32; ++kt) {
    // A (this code's contiguous k order, same as B): lanes 0-15 hold K 0..15, lanes 16-31
    // hold K 16..31; M = lane&15 (tile 0) / +16 (tile 1).
    const int ka = kt * 32 + ((lane < 16) ? 0 : 16);
    const v8h p0 = *(const v8h*)(&s_h[m0][ka]);
    const v8h p1 = *(const v8h*)(&s_h[m0][ka + 8]);
    const v8h q0 = *(const v8h*)(&s_h[m0 + 16][ka]);
    const v8h q1 = *(const v8h*)(&s_h[m0 + 16][ka + 8]);
    const v16h a0 = __builtin_shufflevector(p0, p1, 0, 1, 2, 3, 4, 5, 6, 7,
                                            8, 9, 10, 11, 12, 13, 14, 15);
    const v16h a1 = __builtin_shufflevector(q0, q1, 0, 1, 2, 3, 4, 5, 6, 7,
                                            8, 9, 10, 11, 12, 13, 14, 15);
    // B (same contiguous k order as A): lanes 0-15 K=0..15, lanes 16-31 K=16..31,
    // column N = lane&15. Wt is transposed+padded f16: contiguous 32B load.
    const int kb = kt * 32 + ((lane < 16) ? 0 : 16);
    const v16h bf = *(const v16h*)(wrow + kb);

    acc0 = __builtin_amdgcn_wmma_f32_16x16x32_f16(false, a0, false, bf,
                                                  (short)0, acc0, false, false);
    acc1 = __builtin_amdgcn_wmma_f32_16x16x32_f16(false, a1, false, bf,
                                                  (short)0, acc1, false, false);
  }

  // ---- k-max epilogue: max over 32 rows, then bias + relu ----
  float mx = acc0[0];
  #pragma unroll
  for (int r = 1; r < 8; ++r) mx = fmaxf(mx, acc0[r]);
  #pragma unroll
  for (int r = 0; r < 8; ++r) mx = fmaxf(mx, acc1[r]);
  mx = fmaxf(mx, __shfl_xor(mx, 16)); // combine M halves (same N column)
  if (lane < 16) {
    const float o = mx + bias[col];
    out_feat[((size_t)(b * S + s)) * COUT + col] = fmaxf(o, 0.0f);
  }
}

// ---------------------------------------------------------------------------
// Global max pool over S=128 centers -> (B, 256)
// ---------------------------------------------------------------------------
__global__ void gmaxpool_kernel(const float* __restrict__ feat,
                                float* __restrict__ out) {
  const int b = blockIdx.x, c = threadIdx.x; // 256 threads
  float m = -1e30f;
  for (int s = 0; s < 128; ++s)
    m = fmaxf(m, feat[((size_t)b * 128 + s) * 256 + c]);
  out[b * 256 + c] = m;
}

// ---------------------------------------------------------------------------
// FC head with batch-norm over the 16-row batch. One block, exact f32.
// ---------------------------------------------------------------------------
__global__ void __launch_bounds__(256)
head_kernel(const float* __restrict__ f,
            const float* __restrict__ fc1w, const float* __restrict__ fc1b,
            const float* __restrict__ fc2w, const float* __restrict__ fc2b,
            const float* __restrict__ fc3w, const float* __restrict__ fc3b,
            const float* __restrict__ g1, const float* __restrict__ bb1,
            const float* __restrict__ g2, const float* __restrict__ bb2,
            float* __restrict__ out) {
  __shared__ float X[BATCH * 256];
  __shared__ float Y[BATCH * 256];
  const int t = threadIdx.x;

  for (int e = t; e < BATCH * 256; e += 256) X[e] = f[e];
  __syncthreads();

  float col[BATCH];
  // fc1 (256->256) + bn1 + relu, thread t owns output column t
  {
    for (int b = 0; b < BATCH; ++b) {
      float acc = fc1b[t];
      for (int k = 0; k < 256; ++k) acc += X[b * 256 + k] * fc1w[k * 256 + t];
      col[b] = acc;
    }
    float mean = 0.0f;
    for (int b = 0; b < BATCH; ++b) mean += col[b];
    mean *= (1.0f / BATCH);
    float var = 0.0f;
    for (int b = 0; b < BATCH; ++b) {
      const float d = col[b] - mean;
      var += d * d;
    }
    var *= (1.0f / BATCH);
    const float sc = rsqrtf(var + 1e-5f) * g1[t];
    const float sb = bb1[t];
    for (int b = 0; b < BATCH; ++b)
      Y[b * 256 + t] = fmaxf((col[b] - mean) * sc + sb, 0.0f);
  }
  __syncthreads();

  // fc2 (256->128) + bn2 + relu, threads 0..127
  if (t < 128) {
    for (int b = 0; b < BATCH; ++b) {
      float acc = fc2b[t];
      for (int k = 0; k < 256; ++k) acc += Y[b * 256 + k] * fc2w[k * 128 + t];
      col[b] = acc;
    }
    float mean = 0.0f;
    for (int b = 0; b < BATCH; ++b) mean += col[b];
    mean *= (1.0f / BATCH);
    float var = 0.0f;
    for (int b = 0; b < BATCH; ++b) {
      const float d = col[b] - mean;
      var += d * d;
    }
    var *= (1.0f / BATCH);
    const float sc = rsqrtf(var + 1e-5f) * g2[t];
    const float sb = bb2[t];
    for (int b = 0; b < BATCH; ++b)
      X[b * 128 + t] = fmaxf((col[b] - mean) * sc + sb, 0.0f);
  }
  __syncthreads();

  // fc3 (128->12), threads 0..191: one (b, o) each
  if (t < BATCH * 12) {
    const int b = t / 12, o = t % 12;
    float acc = fc3b[o];
    for (int k = 0; k < 128; ++k) acc += X[b * 128 + k] * fc3w[k * 12 + o];
    out[b * 12 + o] = acc;
  }
}

// ---------------------------------------------------------------------------
extern "C" void kernel_launch(void* const* d_in, const int* in_sizes, int n_in,
                              void* d_out, int out_size, void* d_ws, size_t ws_size,
                              hipStream_t stream) {
  const float* points = (const float*)d_in[0];
  const float* W1 = (const float*)d_in[1];  const float* b1 = (const float*)d_in[2];
  const float* W2 = (const float*)d_in[3];  const float* b2 = (const float*)d_in[4];
  const float* W3 = (const float*)d_in[5];  const float* b3 = (const float*)d_in[6];
  const float* fc1w = (const float*)d_in[7];  const float* fc1b = (const float*)d_in[8];
  const float* fc2w = (const float*)d_in[9];  const float* fc2b = (const float*)d_in[10];
  const float* fc3w = (const float*)d_in[11]; const float* fc3b = (const float*)d_in[12];
  const float* g1 = (const float*)d_in[13];   const float* bb1 = (const float*)d_in[14];
  const float* g2 = (const float*)d_in[15];   const float* bb2 = (const float*)d_in[16];

  float* ws    = (float*)d_ws;
  float* xyz1  = ws;                          // 16*512*3
  float* feat1 = xyz1 + BATCH * 512 * 3;      // 16*512*64
  float* xyz2  = feat1 + BATCH * 512 * 64;    // 16*256*3
  float* feat2 = xyz2 + BATCH * 256 * 3;      // 16*256*128
  float* xyz3  = feat2 + BATCH * 256 * 128;   // 16*128*3
  float* feat3 = xyz3 + BATCH * 128 * 3;      // 16*128*256
  float* gmax  = feat3 + BATCH * 128 * 256;   // 16*256
  _Float16* Wt1 = (_Float16*)(gmax + BATCH * 256); // 64*32 f16
  _Float16* Wt2 = Wt1 + 64 * 32;                   // 128*96 f16
  _Float16* Wt3 = Wt2 + 128 * 96;                  // 256*160 f16

  // Weight prep (tiny, independent of FPS chain)
  prep_weights_kernel<6, 32, 64><<<(64 * 32 + 255) / 256, 256, 0, stream>>>(W1, Wt1);
  prep_weights_kernel<67, 96, 128><<<(128 * 96 + 255) / 256, 256, 0, stream>>>(W2, Wt2);
  prep_weights_kernel<131, 160, 256><<<(256 * 160 + 255) / 256, 256, 0, stream>>>(W3, Wt3);

  // Stage 1: N=4096 -> S=512, feat = xyz (CIN=3, CK=6, KPAD=32, COUT=64)
  fps_kernel<4096, 512, 1024><<<BATCH, 1024, 4096 * 4 * sizeof(float), stream>>>(points, xyz1);
  group_mlp_kernel<4096, 512, 3, 6, 32, 64>
      <<<BATCH * 512, 128, 0, stream>>>(points, points, xyz1, Wt1, b1, feat1, 0.04f);

  // Stage 2: N=512 -> S=256 (CIN=64, CK=67, KPAD=96, COUT=128)
  fps_kernel<512, 256, 512><<<BATCH, 512, 512 * 4 * sizeof(float), stream>>>(xyz1, xyz2);
  group_mlp_kernel<512, 256, 64, 67, 96, 128>
      <<<BATCH * 256, 256, 0, stream>>>(xyz1, feat1, xyz2, Wt2, b2, feat2, 0.16f);

  // Stage 3: N=256 -> S=128 (CIN=128, CK=131, KPAD=160, COUT=256)
  fps_kernel<256, 128, 256><<<BATCH, 256, 256 * 4 * sizeof(float), stream>>>(xyz2, xyz3);
  group_mlp_kernel<256, 128, 128, 131, 160, 256>
      <<<BATCH * 128, 512, 0, stream>>>(xyz2, feat2, xyz3, Wt3, b3, feat3, 0.64f);

  // Global max pool + FC head
  gmaxpool_kernel<<<BATCH, 256, 0, stream>>>(feat3, gmax);
  head_kernel<<<1, 256, 0, stream>>>(gmax, fc1w, fc1b, fc2w, fc2b, fc3w, fc3b,
                                     g1, bb1, g2, bb2, (float*)d_out);
}
